// GAT_75909251990243
// MI455X (gfx1250) — compile-verified, matrix operand order checked by reading, not run on hardware
//
#include <hip/hip_runtime.h>
#include <hip/hip_bf16.h>
#include <math.h>

// Problem constants (from reference)
#define BB 16
#define CIN 2
#define NN 207
#define TT 12
#define EE 1863
#define EMB 64
#define HEADS 8
#define FD 768           // EMB*TT
#define HF 6144          // HEADS*FD
#define MROWS (BB*NN)    // 3312

typedef __attribute__((ext_vector_type(16))) __bf16 v16bf;
typedef __attribute__((ext_vector_type(8)))  float  v8f;

__device__ __forceinline__ unsigned short f2bf(float f) {
    unsigned u = __float_as_uint(f);
    unsigned r = u + 0x7FFFu + ((u >> 16) & 1u);   // round-to-nearest-even
    return (unsigned short)(r >> 16);
}

// ---------------------------------------------------------------------------
// 1) 1x1 convs: xm = (Ws*x + bs) + leaky_relu(Wc*x + bc, 0.01)
//    writes f32 xm into d_out and bf16 h into hb ([B*N, 768], f = c*T + t)
// ---------------------------------------------------------------------------
__global__ __launch_bounds__(256) void conv_kernel(
    const float* __restrict__ x, const float* __restrict__ Ws,
    const float* __restrict__ bs, const float* __restrict__ Wc,
    const float* __restrict__ bc, float* __restrict__ out,
    unsigned short* __restrict__ hb)
{
    const int bn = blockIdx.x;           // b*N + n
    const int b  = bn / NN, n = bn - b * NN;
    for (int f = threadIdx.x; f < FD; f += 256) {
        const int c = f / TT, t = f - c * TT;
        const float xa = x[(((size_t)b * CIN + 0) * NN + n) * TT + t];
        const float xb = x[(((size_t)b * CIN + 1) * NN + n) * TT + t];
        float v1 = Ws[c * 2 + 0] * xa + Ws[c * 2 + 1] * xb + bs[c];
        float v2 = Wc[c * 2 + 0] * xa + Wc[c * 2 + 1] * xb + bc[c];
        v2 = v2 > 0.f ? v2 : 0.01f * v2;
        const float xm = v1 + v2;
        out[(((size_t)b * EMB + c) * NN + n) * TT + t] = xm;
        hb[(size_t)bn * FD + f] = f2bf(xm);
    }
}

// ---------------------------------------------------------------------------
// 2) Deterministic CSR-by-dst build (one block)
// ---------------------------------------------------------------------------
__global__ __launch_bounds__(256) void build_csr(
    const int* __restrict__ dst, int* __restrict__ offs, int* __restrict__ csr)
{
    __shared__ int deg[NN];
    const int tid = threadIdx.x;
    for (int n = tid; n < NN; n += 256) {
        int d = 0;
        for (int e = 0; e < EE; ++e) d += (dst[e] == n);
        deg[n] = d;
    }
    __syncthreads();
    if (tid == 0) {
        int run = 0;
        for (int n = 0; n < NN; ++n) { offs[n] = run; run += deg[n]; }
        offs[NN] = run;
    }
    __syncthreads();
    for (int n = tid; n < NN; n += 256) {
        int cur = offs[n];
        for (int e = 0; e < EE; ++e)
            if (dst[e] == n) csr[cur++] = e;   // ordered by edge id: deterministic
    }
}

// ---------------------------------------------------------------------------
// 3) Weight convert + transpose: W [768, 6144] f32 -> WbT [6144, 768] bf16
// ---------------------------------------------------------------------------
__global__ __launch_bounds__(256) void wconvert_kernel(
    const float* __restrict__ W, unsigned short* __restrict__ WbT)
{
    const size_t idx = (size_t)blockIdx.x * 256 + threadIdx.x;
    if (idx >= (size_t)HF * FD) return;
    const int col = (int)(idx / FD);        // 0..6143
    const int k   = (int)(idx - (size_t)col * FD);
    WbT[idx] = f2bf(W[(size_t)k * HF + col]);
}

// ---------------------------------------------------------------------------
// 4) GEMM: feat[M,6144] = A[M,768](bf16) x W[768,6144] via v_wmma_f32_16x16x32_bf16
//    block = 256 threads (8 waves); wave tile = 16(M) x 64(N); grid = (M/16, 6144/512)
//    Fragment layout per ISA: lanes 0-15 -> M/N idx, K 0-7 & 16-23;
//    lanes 16-31 -> K 8-15 & 24-31 (two contiguous 16B chunks per lane).
//    Software-pipelined: fragments for K-chunk k+1 are loaded while the four
//    WMMAs for chunk k execute (double buffer, unrolled x2, branchless tail
//    prefetch covered by workspace padding).
// ---------------------------------------------------------------------------
union Frag { v16bf v; uint4 q[2]; };

#define LOAD_FRAGS(fa, f0, f1, f2, f3, o)                                     \
    do {                                                                      \
        (fa).q[0] = *(const uint4*)(aRow + (o));                              \
        (fa).q[1] = *(const uint4*)(aRow + (o) + 16);                         \
        (f0).q[0] = *(const uint4*)(bR0 + (o));                               \
        (f0).q[1] = *(const uint4*)(bR0 + (o) + 16);                          \
        (f1).q[0] = *(const uint4*)(bR1 + (o));                               \
        (f1).q[1] = *(const uint4*)(bR1 + (o) + 16);                          \
        (f2).q[0] = *(const uint4*)(bR2 + (o));                               \
        (f2).q[1] = *(const uint4*)(bR2 + (o) + 16);                          \
        (f3).q[0] = *(const uint4*)(bR3 + (o));                               \
        (f3).q[1] = *(const uint4*)(bR3 + (o) + 16);                          \
    } while (0)

#define DO_WMMA(fa, f0, f1, f2, f3)                                           \
    do {                                                                      \
        acc0 = __builtin_amdgcn_wmma_f32_16x16x32_bf16(false, (fa).v, false, (f0).v, (short)0, acc0, false, false); \
        acc1 = __builtin_amdgcn_wmma_f32_16x16x32_bf16(false, (fa).v, false, (f1).v, (short)0, acc1, false, false); \
        acc2 = __builtin_amdgcn_wmma_f32_16x16x32_bf16(false, (fa).v, false, (f2).v, (short)0, acc2, false, false); \
        acc3 = __builtin_amdgcn_wmma_f32_16x16x32_bf16(false, (fa).v, false, (f3).v, (short)0, acc3, false, false); \
    } while (0)

__global__ __launch_bounds__(256) void gemm_bf16_wmma(
    const unsigned short* __restrict__ A,    // [MROWS, 768] bf16 (+pad)
    const unsigned short* __restrict__ BT,   // [6144, 768] bf16 (W^T, +pad)
    float* __restrict__ C)                   // [MROWS, 6144] f32
{
    const int lane = threadIdx.x & 31;
    const int wave = threadIdx.x >> 5;
    const int row0 = blockIdx.x * 16;
    const int col0 = (blockIdx.y * 8 + wave) * 64;
    const int mIn  = lane & 15;
    const int kh   = lane >> 4;              // 0 or 1

    v8f acc0 = {}, acc1 = {}, acc2 = {}, acc3 = {};

    const unsigned short* aRow = A  + (size_t)(row0 + mIn) * FD;
    const unsigned short* bR0  = BT + (size_t)(col0 +  0 + mIn) * FD;
    const unsigned short* bR1  = BT + (size_t)(col0 + 16 + mIn) * FD;
    const unsigned short* bR2  = BT + (size_t)(col0 + 32 + mIn) * FD;
    const unsigned short* bR3  = BT + (size_t)(col0 + 48 + mIn) * FD;

    // Two fragment buffers: X (even chunks) / Y (odd chunks)
    Frag aX, b0X, b1X, b2X, b3X;
    Frag aY, b0Y, b1Y, b2Y, b3Y;

    const int kbase = 8 * kh;
    LOAD_FRAGS(aX, b0X, b1X, b2X, b3X, kbase);          // chunk 0

    // 24 K-chunks of 32, processed two per iteration (12 iterations).
    for (int k0 = 0; k0 < FD; k0 += 64) {
        // prefetch odd chunk (k0+32) while computing even chunk
        LOAD_FRAGS(aY, b0Y, b1Y, b2Y, b3Y, k0 + 32 + kbase);
        DO_WMMA(aX, b0X, b1X, b2X, b3X);
        // prefetch next even chunk (k0+64); final one reads into the tail pad
        LOAD_FRAGS(aX, b0X, b1X, b2X, b3X, k0 + 64 + kbase);
        DO_WMMA(aY, b0Y, b1Y, b2Y, b3Y);
    }

    // C/D layout: lane n = lane&15, VGPR j holds row m = j + 8*(lane>>4)
    const int nIdx = lane & 15;
#pragma unroll
    for (int j = 0; j < 8; ++j) {
        const int m = row0 + j + 8 * kh;
        float* crow = C + (size_t)m * HF;
        crow[col0 +  0 + nIdx] = acc0[j];
        crow[col0 + 16 + nIdx] = acc1[j];
        crow[col0 + 32 + nIdx] = acc2[j];
        crow[col0 + 48 + nIdx] = acc3[j];
    }
}

// ---------------------------------------------------------------------------
// 5) Attention scores: el/er[b,n,h] = sum_f feat[b,n,h,f] * al/ar[h,f]
// ---------------------------------------------------------------------------
__global__ __launch_bounds__(256) void scores_kernel(
    const float* __restrict__ feat, const float* __restrict__ al,
    const float* __restrict__ ar, float* __restrict__ el, float* __restrict__ er)
{
    const int idx = blockIdx.x * 256 + threadIdx.x;     // (b*N+n)*8 + h
    if (idx >= MROWS * HEADS) return;
    const int h = idx & 7, bn = idx >> 3;
    const float* fp  = feat + (size_t)bn * HF + (size_t)h * FD;
    const float* alp = al + h * FD;
    const float* arp = ar + h * FD;
    float sl = 0.f, sr = 0.f;
    for (int f = 0; f < FD; ++f) { const float v = fp[f]; sl += v * alp[f]; sr += v * arp[f]; }
    el[idx] = sl; er[idx] = sr;
}

// ---------------------------------------------------------------------------
// 6) Edge softmax per (b, dst-node, head): alpha = softmax(leaky_relu(el+er,0.2))
// ---------------------------------------------------------------------------
__global__ __launch_bounds__(32) void edge_softmax_kernel(
    const float* __restrict__ el, const float* __restrict__ er,
    const int* __restrict__ srcIdx, const int* __restrict__ offs,
    const int* __restrict__ csr, float* __restrict__ alpha)
{
    const int bn = blockIdx.x;
    const int b = bn / NN, n = bn - b * NN;
    const int h = threadIdx.x;
    if (h >= HEADS) return;
    const float ern = er[bn * HEADS + h];
    const int s0 = offs[n], s1 = offs[n + 1];
    float m = -1e30f;
    for (int i = s0; i < s1; ++i) {
        const int e = csr[i];
        float sc = el[(b * NN + srcIdx[e]) * HEADS + h] + ern;
        sc = sc > 0.f ? sc : 0.2f * sc;
        m = fmaxf(m, sc);
    }
    float den = 0.f;
    for (int i = s0; i < s1; ++i) {
        const int e = csr[i];
        float sc = el[(b * NN + srcIdx[e]) * HEADS + h] + ern;
        sc = sc > 0.f ? sc : 0.2f * sc;
        const float ex = expf(sc - m);
        alpha[((size_t)b * EE + e) * HEADS + h] = ex;
        den += ex;
    }
    const float inv = 1.f / den;
    for (int i = s0; i < s1; ++i) {
        const int e = csr[i];
        alpha[((size_t)b * EE + e) * HEADS + h] *= inv;
    }
}

// ---------------------------------------------------------------------------
// 7) Aggregation: out[b,n,f] = mean_h elu( sum_e alpha*feat[b,src,h,f] + bias[h,f] )
//    mode 0: write bf16 into hb (next layer input). mode 1: add into d_out.
// ---------------------------------------------------------------------------
__global__ __launch_bounds__(256) void aggregate_kernel(
    const float* __restrict__ feat, const float* __restrict__ alpha,
    const float* __restrict__ bias, const int* __restrict__ srcIdx,
    const int* __restrict__ offs, const int* __restrict__ csr,
    unsigned short* __restrict__ hbOut, float* __restrict__ gcOut, int mode)
{
    const int bn = blockIdx.x;
    const int b = bn / NN, n = bn - b * NN;
    const int tid = threadIdx.x;
    const int e0 = offs[n], e1 = offs[n + 1];
    float outv[3] = {0.f, 0.f, 0.f};
    for (int h = 0; h < HEADS; ++h) {
#pragma unroll
        for (int c = 0; c < 3; ++c) {
            const int f = tid + 256 * c;
            float acc = 0.f;
            for (int i = e0; i < e1; ++i) {
                const int e = csr[i];
                const int s = srcIdx[e];
                acc += alpha[((size_t)b * EE + e) * HEADS + h]
                     * feat[((size_t)(b * NN + s)) * HF + (size_t)h * FD + f];
            }
            acc += bias[h * FD + f];
            const float v = acc > 0.f ? acc : expm1f(acc);   // ELU
            outv[c] += v * (1.0f / HEADS);
        }
    }
#pragma unroll
    for (int c = 0; c < 3; ++c) {
        const int f = tid + 256 * c;
        if (mode == 0) {
            hbOut[(size_t)bn * FD + f] = f2bf(outv[c]);
        } else {
            const int ch = f / TT, t = f - ch * TT;
            gcOut[(((size_t)b * EMB + ch) * NN + n) * TT + t] += outv[c];
        }
    }
}

// ---------------------------------------------------------------------------
// Launch
// ---------------------------------------------------------------------------
extern "C" void kernel_launch(void* const* d_in, const int* in_sizes, int n_in,
                              void* d_out, int out_size, void* d_ws, size_t ws_size,
                              hipStream_t stream) {
    const float* x   = (const float*)d_in[0];
    const int*   src = (const int*)  d_in[1];
    const int*   dst = (const int*)  d_in[2];
    const float* Ws  = (const float*)d_in[3];
    const float* bs  = (const float*)d_in[4];
    const float* Wc  = (const float*)d_in[5];
    const float* bc  = (const float*)d_in[6];
    const float* W1  = (const float*)d_in[7];
    const float* al1 = (const float*)d_in[8];
    const float* ar1 = (const float*)d_in[9];
    const float* b1  = (const float*)d_in[10];
    const float* W2  = (const float*)d_in[11];
    const float* al2 = (const float*)d_in[12];
    const float* ar2 = (const float*)d_in[13];
    const float* b2  = (const float*)d_in[14];
    float* out = (float*)d_out;

    // Workspace carve-up (256B aligned). hb / WbT get a 4KB tail pad so the
    // GEMM's branchless last-iteration prefetch stays inside mapped memory.
    char* ws = (char*)d_ws;
    size_t cur = 0;
    auto alloc = [&](size_t bytes) { void* p = ws + cur; cur += (bytes + 255) & ~(size_t)255; return p; };
    unsigned short* hb   = (unsigned short*)alloc((size_t)MROWS * FD * 2 + 4096); // 5.1 MB
    unsigned short* WbT  = (unsigned short*)alloc((size_t)HF * FD * 2 + 4096);    // 9.4 MB
    float*          feat = (float*)alloc((size_t)MROWS * HF * 4);                 // 81.4 MB
    float*          el   = (float*)alloc((size_t)MROWS * HEADS * 4);
    float*          er   = (float*)alloc((size_t)MROWS * HEADS * 4);
    float*          alp  = (float*)alloc((size_t)BB * EE * HEADS * 4);
    int*            offs = (int*)alloc((NN + 1) * 4);
    int*            csr  = (int*)alloc(EE * 4);

    const dim3 gemmGrid(MROWS / 16, HF / 512);    // 207 x 12, 256-thread blocks
    const int wcBlocks = (int)(((size_t)HF * FD + 255) / 256);
    const int scBlocks = (MROWS * HEADS + 255) / 256;

    // xm -> d_out, h(bf16) -> hb ; CSR once
    conv_kernel<<<MROWS, 256, 0, stream>>>(x, Ws, bs, Wc, bc, out, hb);
    build_csr<<<1, 256, 0, stream>>>(dst, offs, csr);

    // ---- GAT layer 1 ----
    wconvert_kernel<<<wcBlocks, 256, 0, stream>>>(W1, WbT);
    gemm_bf16_wmma<<<gemmGrid, 256, 0, stream>>>(hb, WbT, feat);
    scores_kernel<<<scBlocks, 256, 0, stream>>>(feat, al1, ar1, el, er);
    edge_softmax_kernel<<<MROWS, 32, 0, stream>>>(el, er, src, offs, csr, alp);
    aggregate_kernel<<<MROWS, 256, 0, stream>>>(feat, alp, b1, src, offs, csr,
                                                hb, out, /*mode=*/0);

    // ---- GAT layer 2 ----
    wconvert_kernel<<<wcBlocks, 256, 0, stream>>>(W2, WbT);
    gemm_bf16_wmma<<<gemmGrid, 256, 0, stream>>>(hb, WbT, feat);
    scores_kernel<<<scBlocks, 256, 0, stream>>>(feat, al2, ar2, el, er);
    edge_softmax_kernel<<<MROWS, 32, 0, stream>>>(el, er, src, offs, csr, alp);
    aggregate_kernel<<<MROWS, 256, 0, stream>>>(feat, alp, b2, src, offs, csr,
                                                hb, out, /*mode=*/1);  // out += gc
}
